// vit22_tformer_12738873000082
// MI455X (gfx1250) — matrix operand order repaired and checked by reading, not compiled, not run on hardware
//
#include <hip/hip_runtime.h>
#include <math.h>

#define S_LEN 2048
#define D_MODEL 1024
#define N_HEADS 16
#define D_HEAD 64
#define FF_IN 4096
#define FUSED_N 8192

typedef __attribute__((ext_vector_type(16))) __bf16 v16bf;
typedef __attribute__((ext_vector_type(8)))  __bf16 v8bf;
typedef __attribute__((ext_vector_type(8)))  float  v8f;

// ---------------------------------------------------------------------------
// WMMA helpers (CDNA5 16x16x32 bf16, f32 accumulate)
// ---------------------------------------------------------------------------
__device__ inline v8f wmma_bf16(v16bf a, v16bf b, v8f c) {
  return __builtin_amdgcn_wmma_f32_16x16x32_bf16(false, a, false, b, (short)0, c,
                                                 false, false);
}

__device__ inline v8f v8f_zero() {
  v8f r;
#pragma unroll
  for (int i = 0; i < 8; ++i) r[i] = 0.0f;
  return r;
}

// A fragment (16x32 bf16): lane m=lane&15 holds row m; half=lane>>4 selects
// K groups [half*8, half*8+7] and [16+half*8, 16+half*8+7].
__device__ inline v16bf load_afrag(const __bf16* rowbase, int ld) {
  int lane = threadIdx.x & 31;
  int m = lane & 15, hv = lane >> 4;
  const __bf16* p = rowbase + m * ld + hv * 8;
  v8bf lo = *(const v8bf*)p;
  v8bf hi = *(const v8bf*)(p + 16);
  v16bf r;
#pragma unroll
  for (int i = 0; i < 8; ++i) { r[i] = lo[i]; r[i + 8] = hi[i]; }
  return r;
}

// B fragment (32x16 bf16) from an N-major tile (tile[n][k]): lane n=lane&15,
// half=lane>>4 selects K groups [half*8, half*8+7] and [16+half*8, 16+half*8+7].
__device__ inline v16bf load_bfrag(const __bf16* tile, int ld) {
  int lane = threadIdx.x & 31;
  int n = lane & 15, kh = lane >> 4;
  const __bf16* p = tile + n * ld + kh * 8;
  v8bf lo = *(const v8bf*)p;
  v8bf hi = *(const v8bf*)(p + 16);
  v16bf r;
#pragma unroll
  for (int i = 0; i < 8; ++i) { r[i] = lo[i]; r[i + 8] = hi[i]; }
  return r;
}

__device__ inline float redmax16(float v) {
#pragma unroll
  for (int off = 1; off < 16; off <<= 1) v = fmaxf(v, __shfl_xor(v, off, 32));
  return v;
}
__device__ inline float redadd16(float v) {
#pragma unroll
  for (int off = 1; off < 16; off <<= 1) v += __shfl_xor(v, off, 32);
  return v;
}

// ---------------------------------------------------------------------------
// Kernel 1: RMSNorm, fp32 -> bf16
// ---------------------------------------------------------------------------
__global__ void k_rmsnorm(const float* __restrict__ hs, __bf16* __restrict__ xb) {
  int s = blockIdx.x, tid = threadIdx.x;
  const float4* row = (const float4*)(hs + (size_t)s * D_MODEL);
  float4 v = row[tid];
  float a = v.x * v.x + v.y * v.y + v.z * v.z + v.w * v.w;
#pragma unroll
  for (int off = 1; off < 32; off <<= 1) a += __shfl_xor(a, off, 32);
  __shared__ float red[8];
  if ((tid & 31) == 0) red[tid >> 5] = a;
  __syncthreads();
  float tot = 0.f;
#pragma unroll
  for (int wv = 0; wv < 8; ++wv) tot += red[wv];
  float r = rsqrtf(tot * (1.0f / D_MODEL) + 1.1920929e-7f);
  __bf16* o = xb + (size_t)s * D_MODEL + tid * 4;
  o[0] = (__bf16)(v.x * r); o[1] = (__bf16)(v.y * r);
  o[2] = (__bf16)(v.z * r); o[3] = (__bf16)(v.w * r);
}

// ---------------------------------------------------------------------------
// Kernel 2: cast+transpose  W[K][N] fp32 -> Wt[N][K] bf16
// ---------------------------------------------------------------------------
__global__ void k_cast_transpose(const float* __restrict__ W, __bf16* __restrict__ Wt,
                                 int K, int N) {
  __shared__ float t[32][33];
  int tx = threadIdx.x & 31, ty = threadIdx.x >> 5;  // 256 threads: ty 0..7
  int n0 = blockIdx.x * 32, k0 = blockIdx.y * 32;
#pragma unroll
  for (int i = 0; i < 4; ++i)
    t[ty + i * 8][tx] = W[(size_t)(k0 + ty + i * 8) * N + n0 + tx];
  __syncthreads();
#pragma unroll
  for (int i = 0; i < 4; ++i)
    Wt[(size_t)(n0 + ty + i * 8) * K + k0 + tx] = (__bf16)t[tx][ty + i * 8];
}

// ---------------------------------------------------------------------------
// Kernel 3: generic WMMA GEMM  C[M][N] = A[M][K] * Bt[N][K]^T (+bias)
// EPI 0: f32 out.  EPI 1: bf16 out.  EPI 2: f32 out = h*lambda + attn + acc + bias
// ---------------------------------------------------------------------------
#define BM 128
#define BN 128
#define BKK 32
#define LDT 40

template <int EPI>
__global__ __launch_bounds__(256, 2) void k_gemm(
    const __bf16* __restrict__ A, const __bf16* __restrict__ Bt,
    const float* __restrict__ bias, void* __restrict__ Cout,
    int M, int N, int K,
    const float* __restrict__ res_h, const float* __restrict__ res_attn,
    const float* __restrict__ lam) {
  __shared__ __bf16 As[BM * LDT];
  __shared__ __bf16 Bs[BN * LDT];
  int tid = threadIdx.x;
  int wid = tid >> 5;
  int waveN = wid & 1, waveM = wid >> 1;  // 4 (M) x 2 (N) wave grid
  int m0 = blockIdx.y * BM, n0 = blockIdx.x * BN;

  v8f acc[2][4];
#pragma unroll
  for (int i = 0; i < 2; ++i)
#pragma unroll
    for (int j = 0; j < 4; ++j) acc[i][j] = v8f_zero();

  for (int kt = 0; kt < K; kt += BKK) {
#pragma unroll
    for (int s = tid; s < BM * 4; s += 256) {  // 512 16B-slots each tile
      int row = s >> 2, c = s & 3;
      const __bf16* ga = A + (size_t)(m0 + row) * K + kt + c * 8;
      const __bf16* gb = Bt + (size_t)(n0 + row) * K + kt + c * 8;
      *(v8bf*)&As[row * LDT + c * 8] = *(const v8bf*)ga;
      *(v8bf*)&Bs[row * LDT + c * 8] = *(const v8bf*)gb;
      if (kt + BKK < K) {  // prefetch next K tile (global_prefetch_b8)
        __builtin_prefetch(ga + BKK, 0, 1);
        __builtin_prefetch(gb + BKK, 0, 1);
      }
    }
    __syncthreads();
    v16bf af[2], bfg[4];
#pragma unroll
    for (int i = 0; i < 2; ++i)
      af[i] = load_afrag(&As[(waveM * 32 + i * 16) * LDT], LDT);
#pragma unroll
    for (int j = 0; j < 4; ++j)
      bfg[j] = load_bfrag(&Bs[(waveN * 64 + j * 16) * LDT], LDT);
#pragma unroll
    for (int i = 0; i < 2; ++i)
#pragma unroll
      for (int j = 0; j < 4; ++j) acc[i][j] = wmma_bf16(af[i], bfg[j], acc[i][j]);
    __syncthreads();
  }

  int lane = tid & 31, nl = lane & 15, mb = (lane >> 4) * 8;
  float lamv = 0.f;
  if (EPI == 2) lamv = *lam;
#pragma unroll
  for (int i = 0; i < 2; ++i) {
#pragma unroll
    for (int j = 0; j < 4; ++j) {
      int gm0 = m0 + waveM * 32 + i * 16 + mb;
      int gn = n0 + waveN * 64 + j * 16 + nl;
      float bv = bias ? bias[gn] : 0.f;
#pragma unroll
      for (int r = 0; r < 8; ++r) {
        size_t idx = (size_t)(gm0 + r) * N + gn;
        float v = acc[i][j][r] + bv;
        if (EPI == 0) ((float*)Cout)[idx] = v;
        else if (EPI == 1) ((__bf16*)Cout)[idx] = (__bf16)v;
        else ((float*)Cout)[idx] = res_h[idx] * lamv + res_attn[idx] + v;
      }
    }
  }
}

// ---------------------------------------------------------------------------
// Kernel 4: per-head l2norm + RoPE, split qkv f32 -> per-head bf16 [H][S][64]
// ---------------------------------------------------------------------------
__device__ inline void rope_one(const float* __restrict__ src, __bf16* __restrict__ dst,
                                size_t ob, float c, float sn, int lane) {
  float x1 = src[lane], x2 = src[lane + 32];
  float n2 = x1 * x1 + x2 * x2;
#pragma unroll
  for (int off = 1; off < 32; off <<= 1) n2 += __shfl_xor(n2, off, 32);
  float inv = 1.0f / fmaxf(sqrtf(n2), 1e-12f);
  x1 *= inv; x2 *= inv;
  dst[ob + lane] = (__bf16)(x1 * c + x2 * sn);
  dst[ob + lane + 32] = (__bf16)(-x1 * sn + x2 * c);
}

__global__ void k_rope(const float* __restrict__ qkv, __bf16* q_a, __bf16* k_a,
                       __bf16* v_a, __bf16* qb_a, __bf16* kb_a) {
  int s = blockIdx.x;
  int wid = threadIdx.x >> 5, lane = threadIdx.x & 31;
  int h = blockIdx.y * 4 + wid;
  const float* base = qkv + (size_t)s * (5 * D_MODEL) + h * D_HEAD;
  float invf = __expf(-((2.0f * lane) / 64.0f) * logf(1000.0f));
  float fr = (float)s * invf;
  float c = (float)(__bf16)cosf(fr);
  float sn = (float)(__bf16)sinf(fr);
  size_t ob = ((size_t)h * S_LEN + s) * D_HEAD;
  rope_one(base + 0 * D_MODEL, q_a, ob, c, sn, lane);
  rope_one(base + 1 * D_MODEL, k_a, ob, c, sn, lane);
  rope_one(base + 3 * D_MODEL, qb_a, ob, c, sn, lane);
  rope_one(base + 4 * D_MODEL, kb_a, ob, c, sn, lane);
  v_a[ob + lane] = (__bf16)base[2 * D_MODEL + lane];
  v_a[ob + lane + 32] = (__bf16)base[2 * D_MODEL + lane + 32];
}

// ---------------------------------------------------------------------------
// Kernel 5a: sb = 0.125 * rowsum(causal(qb @ kb^T))  -> sb[h][row]  (f32)
// block = 128 threads (4 waves x 16 q-rows), grid (S/64, H)
// ---------------------------------------------------------------------------
#define AKD 72  // padded LDS stride for 64-wide tiles

__global__ __launch_bounds__(128, 1) void k_sbsum(
    const __bf16* __restrict__ qb_a, const __bf16* __restrict__ kb_a,
    float* __restrict__ sb) {
  __shared__ __bf16 Kbs[64 * AKD];
  int tid = threadIdx.x, lane = tid & 31, wid = tid >> 5;
  int nl = lane & 15, mb = (lane >> 4) * 8;
  int h = blockIdx.y;
  int q0 = blockIdx.x * 64;
  int qrow0 = q0 + wid * 16;

  const __bf16* qbrow = qb_a + ((size_t)h * S_LEN + qrow0) * D_HEAD;
  v16bf aqb[2];
#pragma unroll
  for (int c = 0; c < 2; ++c) aqb[c] = load_afrag(qbrow + c * 32, D_HEAD);

  float sba[8];
#pragma unroll
  for (int r = 0; r < 8; ++r) sba[r] = 0.f;

  int nkb = q0 / 64 + 1;
  for (int kb = 0; kb < nkb; ++kb) {
    int kbase = kb * 64;
    const __bf16* kbg = kb_a + ((size_t)h * S_LEN + kbase) * D_HEAD;
#pragma unroll
    for (int s = tid; s < 512; s += 128) {
      int row = s >> 3, c = s & 7;
      *(v8bf*)&Kbs[row * AKD + c * 8] = *(const v8bf*)(kbg + row * 64 + c * 8);
    }
    __syncthreads();
    if (kbase <= qrow0) {
#pragma unroll
      for (int kt = 0; kt < 4; ++kt) {
        v8f s_ = v8f_zero();
#pragma unroll
        for (int c = 0; c < 2; ++c)
          s_ = wmma_bf16(aqb[c], load_bfrag(&Kbs[(kt * 16) * AKD + c * 32], AKD), s_);
        int key = kbase + kt * 16 + nl;
#pragma unroll
        for (int r = 0; r < 8; ++r)
          if (key <= qrow0 + mb + r) sba[r] += s_[r];
      }
    }
    __syncthreads();
  }
#pragma unroll
  for (int r = 0; r < 8; ++r) sba[r] = redadd16(sba[r]);
  if (nl == 0) {
#pragma unroll
    for (int r = 0; r < 8; ++r)
      sb[(size_t)h * S_LEN + qrow0 + mb + r] = 0.125f * sba[r];
  }
}

// ---------------------------------------------------------------------------
// Kernel 5b: causal flash attention, 32-key softmax chunks
// block = 128 threads (4 waves x 16 q-rows), grid (S/64, H)
// ---------------------------------------------------------------------------
#define PLD 40  // padded stride of the 16x32 P tile

__global__ __launch_bounds__(128, 1) void k_attn(
    const __bf16* __restrict__ q_a, const __bf16* __restrict__ k_a,
    const __bf16* __restrict__ v_a, __bf16* __restrict__ y,
    const float* __restrict__ sb, const float* __restrict__ l2ns_p) {
  __shared__ __bf16 Ks[64 * AKD];       // [key][dh]
  __shared__ __bf16 Vt[64 * AKD];       // [dh][key]
  __shared__ __bf16 Pb[4 * 16 * PLD];   // per-wave 16x32 P tile

  int tid = threadIdx.x, lane = tid & 31, wid = tid >> 5;
  int nl = lane & 15, mb = (lane >> 4) * 8;
  int h = blockIdx.y;
  int q0 = blockIdx.x * 64;
  int qrow0 = q0 + wid * 16;

  const __bf16* qrow = q_a + ((size_t)h * S_LEN + qrow0) * D_HEAD;
  v16bf aq[2];
#pragma unroll
  for (int c = 0; c < 2; ++c) aq[c] = load_afrag(qrow + c * 32, D_HEAD);

  v8f o[4];
  float rmax[8], rsum[8];
#pragma unroll
  for (int t = 0; t < 4; ++t) o[t] = v8f_zero();
#pragma unroll
  for (int r = 0; r < 8; ++r) { rmax[r] = -INFINITY; rsum[r] = 0.f; }

  int nkb = q0 / 64 + 1;
  for (int kb = 0; kb < nkb; ++kb) {
    int kbase = kb * 64;
    {  // stage K (row-major) and V transposed
      const __bf16* kg = k_a + ((size_t)h * S_LEN + kbase) * D_HEAD;
#pragma unroll
      for (int s = tid; s < 512; s += 128) {
        int row = s >> 3, c = s & 7;
        *(v8bf*)&Ks[row * AKD + c * 8] = *(const v8bf*)(kg + row * 64 + c * 8);
        if (kb + 1 < nkb) __builtin_prefetch(kg + 4096 + row * 64 + c * 8, 0, 1);
      }
      const __bf16* vg = v_a + ((size_t)h * S_LEN + kbase) * D_HEAD;
      int key = tid & 63, part = tid >> 6;
#pragma unroll
      for (int cj = 0; cj < 4; ++cj) {
        v8bf d = *(const v8bf*)(vg + key * 64 + part * 32 + cj * 8);
#pragma unroll
        for (int j = 0; j < 8; ++j) Vt[(part * 32 + cj * 8 + j) * AKD + key] = d[j];
      }
    }
    __syncthreads();

#pragma unroll
    for (int ck = 0; ck < 2; ++ck) {  // two 32-key chunks
      int keybase = kbase + ck * 32;
      if (keybase <= qrow0) {  // wave-uniform
        bool diag = (keybase + 31 > qrow0);
        v8f sc0 = v8f_zero(), sc1 = v8f_zero();
#pragma unroll
        for (int c = 0; c < 2; ++c) {
          sc0 = wmma_bf16(aq[c], load_bfrag(&Ks[(ck * 32) * AKD + c * 32], AKD), sc0);
          sc1 = wmma_bf16(aq[c], load_bfrag(&Ks[(ck * 32 + 16) * AKD + c * 32], AKD), sc1);
        }
        if (diag) {
          int rowoff = qrow0 + mb - keybase;  // key_local <= rowoff + r is live
#pragma unroll
          for (int r = 0; r < 8; ++r) {
            if (nl > rowoff + r) sc0[r] = -INFINITY;
            if (nl + 16 > rowoff + r) sc1[r] = -INFINITY;
          }
        }
        // online softmax over this 32-key chunk
        float ps[8];
        __bf16* Pw = &Pb[wid * 16 * PLD];
#pragma unroll
        for (int r = 0; r < 8; ++r) {
          float pm = redmax16(fmaxf(sc0[r], sc1[r]));
          float nm = fmaxf(rmax[r], pm);
          float sf = __expf(rmax[r] - nm);
          rmax[r] = nm;
          float p0 = __expf(sc0[r] - nm);
          float p1 = __expf(sc1[r] - nm);
          Pw[(mb + r) * PLD + nl] = (__bf16)p0;
          Pw[(mb + r) * PLD + 16 + nl] = (__bf16)p1;
          ps[r] = p0 + p1;
#pragma unroll
          for (int t = 0; t < 4; ++t) o[t][r] *= sf;
          rsum[r] = rsum[r] * sf + 0.f;  // rescale; add after lane-reduce below
        }
#pragma unroll
        for (int r = 0; r < 8; ++r) rsum[r] += redadd16(ps[r]);
        // O += P @ V  (K = 32 keys of this chunk)
        v16bf pa = load_afrag(Pw, PLD);
#pragma unroll
        for (int ot = 0; ot < 4; ++ot)
          o[ot] = wmma_bf16(pa, load_bfrag(&Vt[(ot * 16) * AKD + ck * 32], AKD), o[ot]);
      }
    }
    __syncthreads();
  }

  float l2ns = *l2ns_p;
  float sbv[8];
#pragma unroll
  for (int r = 0; r < 8; ++r) sbv[r] = sb[(size_t)h * S_LEN + qrow0 + mb + r];
#pragma unroll
  for (int ot = 0; ot < 4; ++ot)
#pragma unroll
    for (int r = 0; r < 8; ++r) {
      float val = l2ns * o[ot][r] / rsum[r] + sbv[r];
      y[(size_t)(qrow0 + mb + r) * D_MODEL + h * D_HEAD + ot * 16 + nl] = (__bf16)val;
    }
}

// ---------------------------------------------------------------------------
// Kernel 6: SwiGLU  ff = silu(u[:,4096:]) * u[:,:4096]
// ---------------------------------------------------------------------------
__global__ void k_swiglu(const __bf16* __restrict__ u, __bf16* __restrict__ ff) {
  int i = blockIdx.x * 256 + threadIdx.x;
  int s = i >> 12, j = i & 4095;
  float xp = (float)u[(size_t)s * FUSED_N + j];
  float g = (float)u[(size_t)s * FUSED_N + FF_IN + j];
  float sg = 1.0f / (1.0f + __expf(-g));
  ff[i] = (__bf16)(xp * g * sg);
}

// ---------------------------------------------------------------------------
// Host launcher
// ---------------------------------------------------------------------------
extern "C" void kernel_launch(void* const* d_in, const int* in_sizes, int n_in,
                              void* d_out, int out_size, void* d_ws, size_t ws_size,
                              hipStream_t stream) {
  (void)in_sizes; (void)n_in; (void)out_size; (void)ws_size;
  const float* hs = (const float*)d_in[0];
  const float* Wq = (const float*)d_in[1];
  const float* Wk = (const float*)d_in[2];
  const float* Wv = (const float*)d_in[3];
  const float* Wqb = (const float*)d_in[4];
  const float* Wkb = (const float*)d_in[5];
  const float* Wo = (const float*)d_in[6];
  const float* bo = (const float*)d_in[7];
  const float* W1 = (const float*)d_in[8];
  const float* b1 = (const float*)d_in[9];
  const float* W2 = (const float*)d_in[10];
  const float* b2 = (const float*)d_in[11];
  const float* lam_p = (const float*)d_in[12];
  const float* l2ns_p = (const float*)d_in[13];
  float* out = (float*)d_out;

  char* w = (char*)d_ws;
  auto alloc = [&](size_t n) { char* p = w; w += (n + 255) & ~(size_t)255; return p; };
  const size_t SD = (size_t)S_LEN * D_MODEL;
  __bf16* x_b = (__bf16*)alloc(SD * 2);
  __bf16* Wcat = (__bf16*)alloc(5ull * D_MODEL * D_MODEL * 2);
  __bf16* Wo_t = (__bf16*)alloc((size_t)D_MODEL * D_MODEL * 2);
  __bf16* W1_t = (__bf16*)alloc((size_t)FUSED_N * D_MODEL * 2);
  __bf16* W2_t = (__bf16*)alloc((size_t)D_MODEL * FF_IN * 2);
  float* qkv = (float*)alloc((size_t)S_LEN * 5 * D_MODEL * 4);
  __bf16* q_a = (__bf16*)alloc(SD * 2);
  __bf16* k_a = (__bf16*)alloc(SD * 2);
  __bf16* v_a = (__bf16*)alloc(SD * 2);
  __bf16* qb_a = (__bf16*)alloc(SD * 2);
  __bf16* kb_a = (__bf16*)alloc(SD * 2);
  __bf16* y_b = (__bf16*)alloc(SD * 2);
  float* attn_o = (float*)alloc(SD * 4);
  float* sb_buf = (float*)alloc((size_t)N_HEADS * S_LEN * 4);
  __bf16* ff_b = (__bf16*)alloc((size_t)S_LEN * FF_IN * 2);
  __bf16* u_b = (__bf16*)qkv;  // alias: qkv (40MB f32) dead after rope; u needs 32MB

  // 1. RMSNorm
  k_rmsnorm<<<S_LEN, 256, 0, stream>>>(hs, x_b);
  // 2. weight cast+transpose to N-major bf16
  k_cast_transpose<<<dim3(32, 32), 256, 0, stream>>>(Wq, Wcat + 0ull * D_MODEL * D_MODEL, D_MODEL, D_MODEL);
  k_cast_transpose<<<dim3(32, 32), 256, 0, stream>>>(Wk, Wcat + 1ull * D_MODEL * D_MODEL, D_MODEL, D_MODEL);
  k_cast_transpose<<<dim3(32, 32), 256, 0, stream>>>(Wv, Wcat + 2ull * D_MODEL * D_MODEL, D_MODEL, D_MODEL);
  k_cast_transpose<<<dim3(32, 32), 256, 0, stream>>>(Wqb, Wcat + 3ull * D_MODEL * D_MODEL, D_MODEL, D_MODEL);
  k_cast_transpose<<<dim3(32, 32), 256, 0, stream>>>(Wkb, Wcat + 4ull * D_MODEL * D_MODEL, D_MODEL, D_MODEL);
  k_cast_transpose<<<dim3(32, 32), 256, 0, stream>>>(Wo, Wo_t, D_MODEL, D_MODEL);
  k_cast_transpose<<<dim3(FUSED_N / 32, 32), 256, 0, stream>>>(W1, W1_t, D_MODEL, FUSED_N);
  k_cast_transpose<<<dim3(32, FF_IN / 32), 256, 0, stream>>>(W2, W2_t, FF_IN, D_MODEL);
  // 3. fused q/k/v/qb/kb projection GEMM (N = 5120)
  k_gemm<0><<<dim3(5 * D_MODEL / BN, S_LEN / BM), 256, 0, stream>>>(
      x_b, Wcat, nullptr, qkv, S_LEN, 5 * D_MODEL, D_MODEL, nullptr, nullptr, nullptr);
  // 4. l2norm + RoPE split
  k_rope<<<dim3(S_LEN, N_HEADS / 4), 128, 0, stream>>>(qkv, q_a, k_a, v_a, qb_a, kb_a);
  // 5a. qb*kb causal rowsum bias
  k_sbsum<<<dim3(S_LEN / 64, N_HEADS), 128, 0, stream>>>(qb_a, kb_a, sb_buf);
  // 5b. causal flash attention
  k_attn<<<dim3(S_LEN / 64, N_HEADS), 128, 0, stream>>>(q_a, k_a, v_a, y_b, sb_buf, l2ns_p);
  // 6. output projection
  k_gemm<0><<<dim3(D_MODEL / BN, S_LEN / BM), 256, 0, stream>>>(
      y_b, Wo_t, bo, attn_o, S_LEN, D_MODEL, D_MODEL, nullptr, nullptr, nullptr);
  // 7. FFN up (fused xp|gate), bf16 out (aliases qkv)
  k_gemm<1><<<dim3(FUSED_N / BN, S_LEN / BM), 256, 0, stream>>>(
      x_b, W1_t, b1, u_b, S_LEN, FUSED_N, D_MODEL, nullptr, nullptr, nullptr);
  // 8. SwiGLU
  k_swiglu<<<(S_LEN * FF_IN) / 256, 256, 0, stream>>>(u_b, ff_b);
  // 9. FFN down + fused final residual epilogue
  k_gemm<2><<<dim3(D_MODEL / BN, S_LEN / BM), 256, 0, stream>>>(
      ff_b, W2_t, b2, out, S_LEN, D_MODEL, FF_IN, hs, attn_o, lam_p);
}
